// PeriodicalPatchMixer_22557168239148
// MI455X (gfx1250) — matrix operand order repaired and checked by reading, not compiled, not run on hardware
//
#include <hip/hip_runtime.h>
#include <hip/hip_bf16.h>
#include <math.h>

// ---------------- problem constants ----------------
#define B_   64
#define FN_  512
#define L_   768
#define C_   (FN_*L_)        // 393216
#define ROWS_ (B_*L_)        // 49152 rows in projection MLP
#define NFREQ 385            // rfft bins
#define NT_DFT 49            // 784/16 column tiles (770 padded to 784)
#define KT_DFT 24            // 768/32 K tiles
#define NKCOL  (NT_DFT*8)    // 392 k slots (incl. pad)
#define BCH    8             // batch chunk for period branches
#define NCHUNK (B_/BCH)
#define NMAX   192           // max n (p=4)
#define POSMAX (NMAX*16)     // 3072

typedef __attribute__((ext_vector_type(16))) __bf16 v16bf;
typedef __attribute__((ext_vector_type(8)))  float  v8f;

struct Ctrl { int p[3]; int n[3]; int off[3]; float w[3]; };

// ---------------- helpers ----------------
__device__ __forceinline__ unsigned short f2bf(float f){
  unsigned u = __float_as_uint(f);
  unsigned r = u + 0x7FFFu + ((u>>16)&1u);   // round-to-nearest-even
  return (unsigned short)(r>>16);
}
__device__ __forceinline__ float bf2f(unsigned short h){
  return __uint_as_float(((unsigned)h)<<16);
}
__device__ __forceinline__ float gelu_exact(float x){
  return 0.5f*x*(1.0f+erff(x*0.7071067811865476f));
}
__device__ __forceinline__ v8f wmma_bf16(v16bf a, v16bf b, v8f c){
  return __builtin_amdgcn_wmma_f32_16x16x32_bf16(false, a, false, b, (short)0, c, false, false);
}

// ---------------- 1) batch-norm stats over axis 0 ----------------
__global__ void bn_stats_k(const float* __restrict__ x,
                           float* __restrict__ meanO, float* __restrict__ rstdO){
  int c = blockIdx.x*blockDim.x + threadIdx.x;
  if (c >= C_) return;
  float s=0.f, s2=0.f;
  for (int b=0;b<B_;++b){ float v = x[(size_t)b*C_ + c]; s += v; s2 += v*v; }
  float m = s*(1.0f/B_);
  float var = s2*(1.0f/B_) - m*m;
  meanO[c]=m; rstdO[c]=rsqrtf(var + 1e-5f);
}

// ---------------- 2) apply BN; emit xn_bf16 (B,FN,L) + xnT f32 (B,L,FN) ----------------
__global__ void bn_apply_transpose_k(const float* __restrict__ x,
    const float* __restrict__ mean, const float* __restrict__ rstd,
    const float* __restrict__ g, const float* __restrict__ bb,
    unsigned short* __restrict__ xnbf, float* __restrict__ xnT){
  __shared__ float tile[32][33];
  int b  = blockIdx.z;
  int f0 = blockIdx.y*32;
  int l0 = blockIdx.x*32;
  int tx = threadIdx.x, ty = threadIdx.y;      // 32 x 8
  for (int s=0;s<4;++s){
    int f = f0 + ty + s*8, l = l0 + tx;
    int c = f*L_ + l;
    float v = (x[(size_t)b*C_ + c] - mean[c])*rstd[c]*g[c] + bb[c];
    xnbf[(size_t)b*C_ + c] = f2bf(v);
    tile[ty+s*8][tx] = v;
  }
  __syncthreads();
  for (int s=0;s<4;++s){
    int l = l0 + ty + s*8, f = f0 + tx;
    xnT[((size_t)b*L_ + l)*FN_ + f] = tile[tx][ty+s*8];
  }
}

// ---------------- 3) DFT basis in WMMA B-fragment layout (bf16) ----------------
__global__ void dft_build_k(unsigned short* __restrict__ dftfrag){
  int idx = blockIdx.x*blockDim.x + threadIdx.x;
  const int total = NT_DFT*KT_DFT*512;
  if (idx >= total) return;
  int nt = idx/(KT_DFT*512);
  int rem = idx - nt*(KT_DFT*512);
  int kt = rem>>9;
  int q  = rem&511;
  int lane = q>>4, i = q&15;
  int c = nt*16 + (lane&15);                      // spectral column (2k | 2k+1)
  int lrow = kt*32 + i + ((lane&16)?16:0);        // time index
  float val = 0.f;
  if (c < 2*NFREQ){
    int k = c>>1;
    int r = (k*lrow) % L_;                        // exact integer phase reduction
    float ang = (float)r * (6.283185307179586f/(float)L_);
    val = (c&1) ? sinf(ang) : cosf(ang);          // sign of im irrelevant for |.|
  }
  dftfrag[idx] = f2bf(val);
}

// ---------------- 4) split weights into hi/lo bf16 B-fragment blobs ----------------
__global__ void weight_frag_k(const float* __restrict__ src, int K, int N,
    unsigned short* __restrict__ hi, unsigned short* __restrict__ lo){
  int idx = blockIdx.x*blockDim.x + threadIdx.x;
  int ktiles = K>>5;
  if (idx >= K*N) return;
  int nt = idx/(ktiles*512);
  int rem = idx - nt*(ktiles*512);
  int kt = rem>>9;
  int q  = rem&511;
  int lane=q>>4, i=q&15;
  int n  = nt*16 + (lane&15);
  int kk = kt*32 + (i&7) + ((lane&16)?8:0) + ((i&8)?16:0);
  float v = src[(size_t)kk*N + n];
  unsigned short h = f2bf(v);
  hi[idx]=h; lo[idx]=f2bf(v - bf2f(h));
}

// ---------------- 5) amplitude spectrum GEMM, double-buffered LDS staging ----------------
__global__ __launch_bounds__(256)
void dft_gemm_k(const unsigned short* __restrict__ xnbf,
    const unsigned short* __restrict__ dftfrag, float* __restrict__ partial){
  __shared__ __attribute__((aligned(16))) unsigned short Atile[2][128*40]; // pad: 40 shorts/row
  __shared__ float stage[8][8];
  int tid = threadIdx.x;
  int w = tid>>5, lane = tid&31;
  int rowbase = blockIdx.x*128;
  int nt = blockIdx.y;

  // staging assignment: thread -> (row, 32B half)
  int r = tid>>1, hh = tid&1;
  const unsigned short* gsrc = xnbf + (size_t)(rowbase+r)*L_ + hh*16;
  unsigned* ldst[2] = { (unsigned*)(Atile[0]) + r*20 + hh*8,
                        (unsigned*)(Atile[1]) + r*20 + hh*8 };
  { // preload kt = 0 into buffer 0
    const uint4* gp = (const uint4*)(gsrc);
    uint4 d0 = gp[0], d1 = gp[1];
    unsigned* dst = ldst[0];
    dst[0]=d0.x; dst[1]=d0.y; dst[2]=d0.z; dst[3]=d0.w;
    dst[4]=d1.x; dst[5]=d1.y; dst[6]=d1.z; dst[7]=d1.w;
  }
  __syncthreads();

  v8f acc = {};
  int row = w*16 + (lane&15);
  int khalf = (lane&16)?8:0;                     // (+16 K)/2
  for (int kt=0; kt<KT_DFT; ++kt){
    // issue next tile's global loads first (overlaps with WMMA below)
    uint4 d0, d1;
    bool have_next = (kt+1 < KT_DFT);
    if (have_next){
      const uint4* gp = (const uint4*)(gsrc + (size_t)(kt+1)*32);
      d0 = gp[0]; d1 = gp[1];
      if (kt+2 < KT_DFT) __builtin_prefetch(gsrc + (size_t)(kt+2)*32, 0, 1);
    }
    // compute current tile
    union { v16bf v; unsigned u[8]; } A;
    const unsigned* au = (const unsigned*)(Atile[kt&1]) + row*20;
    #pragma unroll
    for (int j=0;j<8;++j){
      int kp2 = j + khalf;                         // A-frag contiguous k (16-bit 16x32)
      A.u[j] = au[kp2];
    }
    v16bf Bv = *(const v16bf*)(dftfrag + ((size_t)(nt*KT_DFT + kt)*512 + lane*16));
    acc = wmma_bf16(A.v, Bv, acc);
    // stage next tile into the other buffer
    if (have_next){
      unsigned* dst = ldst[(kt+1)&1];
      dst[0]=d0.x; dst[1]=d0.y; dst[2]=d0.z; dst[3]=d0.w;
      dst[4]=d1.x; dst[5]=d1.y; dst[6]=d1.z; dst[7]=d1.w;
    }
    __syncthreads();                               // single barrier per K-step
  }
  // epilogue: amp = sqrt(re^2+im^2), reduce rows deterministically
  float amp = 0.f;
  #pragma unroll
  for (int j=0;j<8;++j){
    float a = acc[j];
    float b2 = __shfl_xor(a, 1, 32);               // re<->im partner column
    if (!(lane&1)) amp += sqrtf(a*a + b2*b2);
  }
  amp += __shfl_xor(amp, 16, 32);                  // combine row halves
  if (((lane&1)==0) && lane<16) stage[w][lane>>1] = amp;
  __syncthreads();
  if (tid<8){
    float s=0.f;
    #pragma unroll
    for (int q=0;q<8;++q) s += stage[q][tid];
    partial[(size_t)blockIdx.x*NKCOL + nt*8 + tid] = s;
  }
}

// ---------------- 6) reduce partials -> freq ----------------
__global__ void freq_reduce_k(const float* __restrict__ partial, float* __restrict__ freq){
  int k = threadIdx.x;
  if (k >= NKCOL) return;
  float s=0.f;
  for (int r=0;r<256;++r) s += partial[(size_t)r*NKCOL + k];
  if (k==0) s = 0.f;
  if (k>=NFREQ) s = -1.0f;
  freq[k]=s;
}

// ---------------- 7) top-k + period logic + softmax fusion weights ----------------
__global__ void select_k(const float* __restrict__ freq, const float* __restrict__ fusion_w,
                         Ctrl* __restrict__ c){
  if (threadIdx.x!=0 || blockIdx.x!=0) return;
  int pick[3]={-1,-1,-1};
  for (int t=0;t<3;++t){
    float best=-1e30f; int bi=-1;
    for (int k=0;k<NFREQ;++k){
      if (k==pick[0]||k==pick[1]||k==pick[2]) continue;
      float v=freq[k];
      if (v>best){best=v;bi=k;}                    // ties -> lowest index (stable)
    }
    pick[t]=bi;
  }
  int per[6]; int np=0;
  for (int t=0;t<3;++t){
    int i=pick[t];
    if (i>0){
      int p=L_/i;
      if (p>0){
        if (p<4) p=4;
        if (p>L_/2) p=L_/2;
        per[np++]=p;
      }
    }
  }
  if (np==0){ per[0]=L_/4; per[1]=L_/8; per[2]=L_/16; np=3; }
  else if (np<3){
    int defs[3]={L_/4,L_/8,L_/16};
    for (int d=0; d<3 && np<3; ++d){
      bool dup=false;
      for (int q=0;q<np;++q) if (per[q]==defs[d]) dup=true;
      if(!dup) per[np++]=defs[d];
    }
  }
  for (int k=0;k<3;++k){
    int p=per[k];
    int n=(L_-p)/p + 1;
    c->p[k]=p; c->n[k]=n; c->off[k]=L_-p*n;
  }
  float f0=fusion_w[0], f1=fusion_w[1], f2=fusion_w[2];
  float mx=fmaxf(f0,fmaxf(f1,f2));
  float e0=expf(f0-mx), e1=expf(f1-mx), e2=expf(f2-mx);
  float s=e0+e1+e2;
  c->w[0]=e0/s; c->w[1]=e1/s; c->w[2]=e2/s;
}

// ---------------- 8) period branch: resize(p->16) + tiny MLP + gelu, writes flat (bl,pos,f) ----------------
__global__ __launch_bounds__(256)
void mixer_k(const float* __restrict__ xnT, const float* __restrict__ W1,
    const float* __restrict__ b1, const float* __restrict__ W2, const float* __restrict__ b2,
    const Ctrl* __restrict__ ctrl, float* __restrict__ flat, int kper, int chunk){
  __shared__ float wl[1072];                       // W1(512) b1(32) W2(512) b2(16)
  int tid=threadIdx.x;
  for (int i=tid;i<1072;i+=256){
    float v;
    if (i<512) v=W1[i];
    else if (i<544) v=b1[i-512];
    else if (i<1056) v=W2[i-544];
    else v=b2[i-1056];
    wl[i]=v;
  }
  __syncthreads();
  int p = ctrl->p[kper], n = ctrl->n[kper], off = ctrl->off[kper];
  int bl = blockIdx.x / NMAX;
  int j  = blockIdx.x % NMAX;
  if (j >= n) return;                              // block-uniform
  int b = chunk*BCH + bl;
  float scale = (float)p * (1.0f/16.0f);
  for (int rr=0; rr<2; ++rr){
    int f = tid + rr*256;
    float xb[16];
    #pragma unroll
    for (int u=0;u<16;++u){
      float pos = ((float)u + 0.5f)*scale - 0.5f;
      pos = fminf(fmaxf(pos, 0.0f), (float)(p-1));
      int lo = (int)floorf(pos);
      int hi = min(lo+1, p-1);
      float t = pos - (float)lo;
      float vA = xnT[((size_t)b*L_ + (off + j*p + lo))*FN_ + f];
      float vB = xnT[((size_t)b*L_ + (off + j*p + hi))*FN_ + f];
      xb[u] = vA*(1.0f-t) + vB*t;
    }
    float h[32];
    #pragma unroll
    for (int q=0;q<32;++q){
      float s = wl[512+q];
      #pragma unroll
      for (int u=0;u<16;++u) s += xb[u]*wl[u*32+q];
      h[q] = gelu_exact(s);
    }
    #pragma unroll
    for (int u=0;u<16;++u){
      float s = wl[1056+u];
      #pragma unroll
      for (int q=0;q<32;++q) s += h[q]*wl[544 + q*16 + u];
      flat[((size_t)bl*POSMAX + (j*16+u))*FN_ + f] = gelu_exact(s);
    }
  }
}

// ---------------- 9) resize (n*16 -> L) + weighted accumulate into fused (B,L,FN) ----------------
__global__ void resize_accum_k(const float* __restrict__ flat, const Ctrl* __restrict__ ctrl,
    float* __restrict__ fused, int kper, int chunk){
  int f  = threadIdx.x;                            // 512 threads, coalesced on f
  int bl = blockIdx.x / L_;
  int l  = blockIdx.x % L_;
  int n = ctrl->n[kper];
  float wk = ctrl->w[kper];
  int P = n*16;
  float pos = ((float)l + 0.5f)*((float)P/(float)L_) - 0.5f;
  pos = fminf(fmaxf(pos,0.0f),(float)(P-1));
  int lo=(int)floorf(pos);
  int hi=min(lo+1,P-1);
  float t = pos-(float)lo;
  float v = flat[((size_t)bl*POSMAX+lo)*FN_+f]*(1.0f-t)
          + flat[((size_t)bl*POSMAX+hi)*FN_+f]*t;
  size_t o = ((size_t)(chunk*BCH+bl)*L_ + l)*FN_ + f;
  if (kper==0) fused[o] = wk*v; else fused[o] += wk*v;
}

// ---------------- 10) fused projection MLP: gelu(A@Wp1+b)@Wp2+b, hi/lo bf16 split (3 WMMAs/tile) ----------------
__global__ __launch_bounds__(256)
void mlp_k(const float* __restrict__ fused,
    const unsigned short* __restrict__ w1hi, const unsigned short* __restrict__ w1lo,
    const float* __restrict__ bp1,
    const unsigned short* __restrict__ w2hi, const unsigned short* __restrict__ w2lo,
    const float* __restrict__ bp2,
    const float* __restrict__ x, float* __restrict__ outpre){
  __shared__ __attribute__((aligned(16))) unsigned short Ahi[16*512];  // A-frag swizzled
  __shared__ __attribute__((aligned(16))) unsigned short Alo[16*512];
  __shared__ __attribute__((aligned(16))) unsigned short Hhi[32*512];  // hidden (K=1024)
  __shared__ __attribute__((aligned(16))) unsigned short Hlo[32*512];
  int tid = threadIdx.x;
  int w = tid>>5, lane = tid&31;
  int row0 = blockIdx.x*16;
  // phase 0: load 16x512 f32 rows, split to hi/lo bf16, store A-frag swizzled
  for (int idx=tid; idx<16*512; idx+=256){
    int m = idx>>9, K = idx&511;
    float v = fused[(size_t)(row0+m)*FN_ + K];
    unsigned short h = f2bf(v);
    unsigned short l2 = f2bf(v - bf2f(h));
    int kt = K>>5, kp = K&31;
    int lane2 = m + ((kp&8)?16:0);
    int i2 = (kp&7) + ((kp&16)?8:0);
    int o = (kt*32+lane2)*16 + i2;
    Ahi[o]=h; Alo[o]=l2;
  }
  __syncthreads();
  // phase 1: H = gelu(A @ Wp1 + bp1); 8 waves x 8 col tiles, K=512
  for (int t=0;t<8;++t){
    int nt1 = w*8+t;
    v8f acc = {};
    for (int kt=0;kt<16;++kt){
      v16bf ah = *(const v16bf*)(&Ahi[(kt*32+lane)*16]);
      v16bf al = *(const v16bf*)(&Alo[(kt*32+lane)*16]);
      size_t bo = ((size_t)(nt1*16+kt)*512 + lane*16);
      if (kt+1<16){                                  // stream next weight frags into GL2
        __builtin_prefetch(w1hi + bo + 512, 0, 1);
        __builtin_prefetch(w1lo + bo + 512, 0, 1);
      }
      v16bf bh  = *(const v16bf*)(w1hi + bo);
      v16bf bl2 = *(const v16bf*)(w1lo + bo);
      acc = wmma_bf16(ah,bh,acc);
      acc = wmma_bf16(ah,bl2,acc);
      acc = wmma_bf16(al,bh,acc);
    }
    int n = nt1*16 + (lane&15);
    float bb = bp1[n];
    int kt2 = n>>5, kp = n&31;
    int i2 = (kp&7)+((kp&16)?8:0);
    #pragma unroll
    for (int j=0;j<8;++j){
      int m = j + ((lane&16)?8:0);
      float hv = gelu_exact(acc[j]+bb);
      unsigned short h = f2bf(hv);
      unsigned short l2 = f2bf(hv - bf2f(h));
      int lane2 = m + ((kp&8)?16:0);
      int o = (kt2*32+lane2)*16+i2;
      Hhi[o]=h; Hlo[o]=l2;
    }
  }
  __syncthreads();
  // phase 2: out = H @ Wp2 + bp2; residual add vs x (transposed); 8 waves x 4 col tiles, K=1024
  int b  = row0 / L_;
  int l0 = row0 % L_;
  for (int t=0;t<4;++t){
    int nt2 = w*4+t;
    v8f acc = {};
    for (int kt=0;kt<32;++kt){
      v16bf ah = *(const v16bf*)(&Hhi[(kt*32+lane)*16]);
      v16bf al = *(const v16bf*)(&Hlo[(kt*32+lane)*16]);
      size_t bo = ((size_t)(nt2*32+kt)*512 + lane*16);
      if (kt+1<32){
        __builtin_prefetch(w2hi + bo + 512, 0, 1);
        __builtin_prefetch(w2lo + bo + 512, 0, 1);
      }
      v16bf bh  = *(const v16bf*)(w2hi + bo);
      v16bf bl2 = *(const v16bf*)(w2lo + bo);
      acc = wmma_bf16(ah,bh,acc);
      acc = wmma_bf16(ah,bl2,acc);
      acc = wmma_bf16(al,bh,acc);
    }
    int f = nt2*16 + (lane&15);
    float bb = bp2[f];
    #pragma unroll
    for (int j=0;j<8;++j){
      int m = j + ((lane&16)?8:0);
      size_t o = (size_t)b*C_ + (size_t)f*L_ + (l0+m);
      outpre[o] = x[o] + acc[j] + bb;
    }
  }
}

// ---------------- 11) final BN apply ----------------
__global__ void bn_apply_out_k(const float* __restrict__ outpre, const float* __restrict__ mean,
    const float* __restrict__ rstd, const float* __restrict__ g, const float* __restrict__ bb,
    float* __restrict__ out){
  size_t i = (size_t)blockIdx.x*blockDim.x + threadIdx.x;
  if (i >= (size_t)B_*C_) return;
  int c = (int)(i % C_);
  out[i] = (outpre[i]-mean[c])*rstd[c]*g[c] + bb[c];
}

// ---------------- host launcher ----------------
extern "C" void kernel_launch(void* const* d_in, const int* in_sizes, int n_in,
                              void* d_out, int out_size, void* d_ws, size_t ws_size,
                              hipStream_t stream){
  (void)in_sizes; (void)n_in; (void)out_size;
  const float* x    = (const float*)d_in[0];
  const float* g_in = (const float*)d_in[1];
  const float* b_in = (const float*)d_in[2];
  const float* W1   = (const float*)d_in[3];
  const float* b1   = (const float*)d_in[4];
  const float* W2   = (const float*)d_in[5];
  const float* b2   = (const float*)d_in[6];
  const float* fw   = (const float*)d_in[7];
  const float* Wp1  = (const float*)d_in[8];
  const float* bp1  = (const float*)d_in[9];
  const float* Wp2  = (const float*)d_in[10];
  const float* bp2  = (const float*)d_in[11];
  const float* g_out= (const float*)d_in[12];
  const float* b_out= (const float*)d_in[13];
  float* out = (float*)d_out;
  char* ws = (char*)d_ws;

  size_t off=0;
  auto alloc=[&](size_t bytes)->size_t{ size_t o=off; off=(off+bytes+255)&~(size_t)255; return o; };
  size_t o_ctrl = alloc(sizeof(Ctrl));
  size_t o_mean0= alloc((size_t)C_*4);
  size_t o_rstd0= alloc((size_t)C_*4);
  size_t o_mean1= alloc((size_t)C_*4);
  size_t o_rstd1= alloc((size_t)C_*4);
  size_t o_freq = alloc((size_t)NKCOL*4);
  size_t o_part = alloc((size_t)256*NKCOL*4);
  size_t o_dft  = alloc((size_t)NT_DFT*KT_DFT*512*2);
  size_t o_w1hi = alloc((size_t)512*1024*2);
  size_t o_w1lo = alloc((size_t)512*1024*2);
  size_t o_w2hi = alloc((size_t)1024*512*2);
  size_t o_w2lo = alloc((size_t)1024*512*2);
  size_t o_xnT  = alloc((size_t)B_*C_*4);   // reused as outpre in the MLP phase
  size_t o_xnbf = alloc((size_t)B_*C_*2);
  size_t o_flat = alloc((size_t)BCH*POSMAX*FN_*4);
  size_t o_fused= alloc((size_t)B_*C_*4);
  if (ws_size < off) return;                 // workspace too small: bail safely

  Ctrl*  ctrl  = (Ctrl*) (ws+o_ctrl);
  float* mean0 = (float*)(ws+o_mean0);
  float* rstd0 = (float*)(ws+o_rstd0);
  float* mean1 = (float*)(ws+o_mean1);
  float* rstd1 = (float*)(ws+o_rstd1);
  float* freq  = (float*)(ws+o_freq);
  float* part  = (float*)(ws+o_part);
  unsigned short* dftf = (unsigned short*)(ws+o_dft);
  unsigned short* w1hi = (unsigned short*)(ws+o_w1hi);
  unsigned short* w1lo = (unsigned short*)(ws+o_w1lo);
  unsigned short* w2hi = (unsigned short*)(ws+o_w2hi);
  unsigned short* w2lo = (unsigned short*)(ws+o_w2lo);
  float* xnT   = (float*)(ws+o_xnT);
  float* outpre= (float*)(ws+o_xnT);         // alias: xnT dead before MLP writes
  unsigned short* xnbf = (unsigned short*)(ws+o_xnbf);
  float* flat  = (float*)(ws+o_flat);
  float* fused = (float*)(ws+o_fused);

  // phase A: BN + spectrum
  bn_stats_k<<<C_/256,256,0,stream>>>(x, mean0, rstd0);
  bn_apply_transpose_k<<<dim3(L_/32, FN_/32, B_), dim3(32,8), 0, stream>>>(
      x, mean0, rstd0, g_in, b_in, xnbf, xnT);
  dft_build_k<<<(NT_DFT*KT_DFT*512)/256,256,0,stream>>>(dftf);
  weight_frag_k<<<(512*1024)/256,256,0,stream>>>(Wp1, 512, 1024, w1hi, w1lo);
  weight_frag_k<<<(1024*512)/256,256,0,stream>>>(Wp2, 1024, 512, w2hi, w2lo);
  dft_gemm_k<<<dim3(256, NT_DFT),256,0,stream>>>(xnbf, dftf, part);
  freq_reduce_k<<<1,512,0,stream>>>(part, freq);
  select_k<<<1,32,0,stream>>>(freq, fw, ctrl);

  // phase B: period branches (worst-case grids; sizes read from device ctrl)
  for (int k=0;k<3;++k){
    for (int ch=0; ch<NCHUNK; ++ch){
      mixer_k<<<BCH*NMAX,256,0,stream>>>(xnT, W1, b1, W2, b2, ctrl, flat, k, ch);
      resize_accum_k<<<BCH*L_,512,0,stream>>>(flat, ctrl, fused, k, ch);
    }
  }

  // phase C: fused projection MLP (WMMA) + residual, then output BN
  mlp_k<<<ROWS_/16,256,0,stream>>>(fused, w1hi, w1lo, bp1, w2hi, w2lo, bp2, x, outpre);
  bn_stats_k<<<C_/256,256,0,stream>>>(outpre, mean1, rstd1);
  bn_apply_out_k<<<(unsigned)(((size_t)B_*C_+255)/256),256,0,stream>>>(
      outpre, mean1, rstd1, g_out, b_out, out);
}
